// SNN_41609643163976
// MI455X (gfx1250) — matrix operand order repaired and checked by reading, not compiled, not run on hardware
//
#include <hip/hip_runtime.h>

typedef __attribute__((ext_vector_type(16))) _Float16 v16h;
typedef __attribute__((ext_vector_type(8)))  _Float16 v8h;
typedef __attribute__((ext_vector_type(8)))  float    v8f;

#define B_   1024
#define T_   64
#define I_   512
#define H_   300
#define HP_  320      // padded hidden (multiple of 32 for WMMA K, of 16 for N)
#define O_   256
#define BT_  (B_ * T_)
#define THRESH 1.0f
#define BETA1  0.9f

// ---------------- weight conversion (fp32 -> f16, zero-padded) ----------------

__global__ __launch_bounds__(256) void cvt_w1(const float* __restrict__ W1,
                                              _Float16* __restrict__ W1h) {
    int i = blockIdx.x * 256 + threadIdx.x;          // over HP_*I_
    if (i >= HP_ * I_) return;
    int row = i / I_, col = i % I_;
    W1h[i] = (row < H_) ? (_Float16)W1[row * I_ + col] : (_Float16)0.0f;
}

__global__ __launch_bounds__(256) void cvt_w2(const float* __restrict__ W2,
                                              _Float16* __restrict__ W2h) {
    int i = blockIdx.x * 256 + threadIdx.x;          // over O_*HP_
    if (i >= O_ * HP_) return;
    int row = i / HP_, col = i % HP_;
    W2h[i] = (col < H_) ? (_Float16)W2[row * H_ + col] : (_Float16)0.0f;
}

// ---------------- WMMA GEMM: C[M,NCOLS] = A[M,K] * W[NCOLS,K]^T ----------------
// Block = 256 threads = 8 waves. Block tile: 64 rows x NCOLS cols.
// Wave (w>>1) picks one of four 16-row stripes; (w&1) picks N half; each wave
// owns NT tiles of 16 columns, f32 accum in registers, K-loop step 32.
// K is a compile-time constant so k-step offsets fold into the 24-bit
// immediate offset of global_load_b128. B-tile loads are issued GRP tiles at
// a time (one clause of 2*GRP b128 loads) before their GRP WMMAs, to keep
// more per-wave loads in flight instead of a wait-0 before every WMMA.

template <int NT, int GRP, int NCOLS, int K, bool A_F32>
__global__ __launch_bounds__(256) void wmma_gemm(const void* __restrict__ Aptr,
                                                 const _Float16* __restrict__ W,
                                                 float* __restrict__ C) {
    const int lane   = threadIdx.x & 31;
    const int w      = threadIdx.x >> 5;
    const int m0     = blockIdx.x * 64 + (w >> 1) * 16;
    const int n0     = (w & 1) * (NT * 16);
    const int cshift = lane >> 4;           // 0 for lanes 0-15, 1 for 16-31
    const int r      = m0 + (lane & 15);    // A row this lane feeds
    const int csel   = lane & 15;

    const float*    apf = (const float*)   Aptr + (size_t)r * K + cshift * 8;
    const _Float16* aph = (const _Float16*)Aptr + (size_t)r * K + cshift * 8;
    // single B base; tile offsets (t*16*K halfs) are compile-time immediates
    const _Float16* bbase = W + (size_t)(n0 + csel) * K + cshift * 8;

    v8f acc[NT];
#pragma unroll
    for (int t = 0; t < NT; ++t)
#pragma unroll
        for (int g = 0; g < 8; ++g) acc[t][g] = 0.0f;

#pragma unroll 2
    for (int kb = 0; kb < K; kb += 32) {
        // ---- A fragment: halves 0-7 = K[koff..+7], halves 8-15 = K[koff+16..+23]
        v16h a;
        if (A_F32) {
            const float4* ap4 = (const float4*)(apf + kb);
            float4 x0 = ap4[0], x1 = ap4[1], x2 = ap4[4], x3 = ap4[5];
            a[0]=(_Float16)x0.x; a[1]=(_Float16)x0.y; a[2]=(_Float16)x0.z; a[3]=(_Float16)x0.w;
            a[4]=(_Float16)x1.x; a[5]=(_Float16)x1.y; a[6]=(_Float16)x1.z; a[7]=(_Float16)x1.w;
            a[8]=(_Float16)x2.x; a[9]=(_Float16)x2.y; a[10]=(_Float16)x2.z; a[11]=(_Float16)x2.w;
            a[12]=(_Float16)x3.x; a[13]=(_Float16)x3.y; a[14]=(_Float16)x3.z; a[15]=(_Float16)x3.w;
        } else {
            const v8h* ap8 = (const v8h*)(aph + kb);
            v8h h0 = ap8[0], h1 = ap8[2];   // +0..7  and +16..23 halves
#pragma unroll
            for (int j = 0; j < 8; ++j) { a[j] = h0[j]; a[8 + j] = h1[j]; }
        }
        // ---- B fragments + WMMA, grouped for memory-level parallelism
#pragma unroll
        for (int tg = 0; tg < NT; tg += GRP) {
            v8h b0[GRP], b1[GRP];
#pragma unroll
            for (int j = 0; j < GRP; ++j) {
                const v8h* bpt =
                    (const v8h*)(bbase + (size_t)(tg + j) * 16 * K + kb);
                b0[j] = bpt[0];
                b1[j] = bpt[2];
            }
#pragma unroll
            for (int j = 0; j < GRP; ++j) {
                v16h b;
#pragma unroll
                for (int e = 0; e < 8; ++e) { b[e] = b0[j][e]; b[8 + e] = b1[j][e]; }
                acc[tg + j] = __builtin_amdgcn_wmma_f32_16x16x32_f16(
                    false, a, false, b, (short)0, acc[tg + j], false, false);
            }
        }
    }
    // ---- store: VGPR g -> row m0 + g + 8*cshift, col n0 + (lane&15)
    const int rowbase = m0 + (cshift << 3);
#pragma unroll
    for (int t = 0; t < NT; ++t) {
        const int col = n0 + t * 16 + csel;
#pragma unroll
        for (int g = 0; g < 8; ++g)
            C[(size_t)(rowbase + g) * NCOLS + col] = acc[t][g];
    }
}

// ---------------- layer-1 LIF scan: CUR1 -> binary SPK1 (f16) ----------------

__global__ __launch_bounds__(256) void lif_scan1(const float* __restrict__ CUR1,
                                                 const float* __restrict__ b1,
                                                 _Float16* __restrict__ SPK1) {
    int i = blockIdx.x * 256 + threadIdx.x;          // over B_*HP_
    if (i >= B_ * HP_) return;
    int b = i / HP_, h = i % HP_;
    float bias = (h < H_) ? b1[h] : 0.0f;
    size_t base = (size_t)b * T_ * HP_ + h;
    float mem = 0.0f;
#pragma unroll 4
    for (int t = 0; t < T_; ++t) {
        float cur   = CUR1[base + (size_t)t * HP_] + bias;
        float reset = (mem > THRESH) ? THRESH : 0.0f;
        mem = BETA1 * mem + cur - reset;
        SPK1[base + (size_t)t * HP_] = (_Float16)((mem > THRESH) ? 1.0f : 0.0f);
    }
}

// -------- layer-2 LIF scan: CUR2 (in d_out spk region) -> spk2 + mem2_rec ----

__global__ __launch_bounds__(256) void lif_scan2(float* __restrict__ out,
                                                 const float* __restrict__ b2,
                                                 const float* __restrict__ beta2) {
    int i = blockIdx.x * 256 + threadIdx.x;          // over B_*O_
    if (i >= B_ * O_) return;
    int b = i / O_, o = i % O_;
    float bias = b2[o];
    float bet  = beta2[o];
    float* memrec = out + (size_t)BT_ * O_;
    float mem = 0.0f;
#pragma unroll 4
    for (int t = 0; t < T_; ++t) {
        size_t p = ((size_t)b * T_ + t) * O_ + o;    // [B,T,O] position
        float cur   = out[p] + bias;                  // CUR2 staged here by GEMM2
        float reset = (mem > THRESH) ? THRESH : 0.0f;
        mem = bet * mem + cur - reset;
        out[p] = (mem > THRESH) ? 1.0f : 0.0f;        // spk2 [B,T,O]
        memrec[((size_t)t * B_ + b) * O_ + o] = mem;  // mem2  [T,B,O]
    }
}

// ------------------------------------------------------------------------------

extern "C" void kernel_launch(void* const* d_in, const int* in_sizes, int n_in,
                              void* d_out, int out_size, void* d_ws, size_t ws_size,
                              hipStream_t stream) {
    const float* x     = (const float*)d_in[0];   // [B,T,I] == [BT, I]
    const float* W1    = (const float*)d_in[1];   // [300, 512]
    const float* b1    = (const float*)d_in[2];   // [300]
    const float* W2    = (const float*)d_in[3];   // [256, 300]
    const float* b2    = (const float*)d_in[4];   // [256]
    const float* beta2 = (const float*)d_in[5];   // [256]
    float* out = (float*)d_out;

    char* ws = (char*)d_ws;
    _Float16* W1h  = (_Float16*)(ws);                                 // 320*512*2
    _Float16* W2h  = (_Float16*)(ws + 327680);                        // 256*320*2
    float*    CUR1 = (float*)   (ws + 327680 + 163840);               // 65536*320*4
    _Float16* SPK1 = (_Float16*)(ws + 327680 + 163840 + 83886080);    // 65536*320*2

    // 1) convert/pad weights to f16
    cvt_w1<<<(HP_ * I_ + 255) / 256, 256, 0, stream>>>(W1, W1h);
    cvt_w2<<<(O_ * HP_ + 255) / 256, 256, 0, stream>>>(W2, W2h);

    // 2) CUR1 = X * W1h^T   (M=65536, N=320, K=512)
    wmma_gemm<10, 5, HP_, I_, true><<<BT_ / 64, 256, 0, stream>>>(x, W1h, CUR1);

    // 3) layer-1 LIF scan -> binary spikes (f16)
    lif_scan1<<<(B_ * HP_ + 255) / 256, 256, 0, stream>>>(CUR1, b1, SPK1);

    // 4) CUR2 = SPK1 * W2h^T staged directly into d_out spk region
    //    (M=65536, N=256, K=320)
    wmma_gemm<8, 4, O_, HP_, false><<<BT_ / 64, 256, 0, stream>>>(SPK1, W2h, out);

    // 5) layer-2 LIF scan in place: spk2 [B,T,O] + mem2_rec [T,B,O]
    lif_scan2<<<(B_ * O_ + 255) / 256, 256, 0, stream>>>(out, b2, beta2);
}
